// LocalMHA_70652212019539
// MI455X (gfx1250) — matrix operand order repaired and checked by reading, not compiled, not run on hardware
//
#include <hip/hip_runtime.h>
#include <cstdint>

// ---------------------------------------------------------------------------
// LocalMHA for MI455X (gfx1250): bf16 WMMA 16x16x32, fp32 accumulate.
// K0: fp32 -> bf16 pre-convert of x / w_qkv / w_out (native cvt, vectorized)
// K1: QKV projection GEMM (all-bf16 operands, scale folded into q)
// K2: windowed attention; k-tile staged via global_load_async_to_lds_b128,
//     scores in 192KB LDS, v^T in 48KB LDS, k-tile in 48KB LDS (288KB total)
// K3: output projection GEMM (bf16 operands, fp32 out)
// ---------------------------------------------------------------------------

typedef __attribute__((ext_vector_type(16))) __bf16 v16bf;
typedef __attribute__((ext_vector_type(8)))  float  v8f;

#define B_   4
#define NSEQ 8192
#define DIM  512
#define NH   8
#define DH   64
#define WIN  128
#define NW   (NSEQ / WIN)   /* 64  */
#define KW   (3 * WIN)      /* 384 */

// K2 dynamic-LDS byte offsets (single dynamic allocation => base offset 0).
#define LDS_SCORES 0                          /* 128*384 f32  = 196608 B */
#define LDS_VT     (WIN * KW * 4)             /* 64*384 bf16  =  49152 B */
#define LDS_KT     (LDS_VT + DH * KW * 2)     /* 384*64 bf16  =  49152 B */
#define LDS_TOTAL  (LDS_KT + KW * DH * 2)     /* 294912 B = 288 KB       */

union Frag {
  v16bf v;
  unsigned u[8];
};

// Native fp32 -> bf16 (RNE via hardware cvt).
__device__ __forceinline__ unsigned short f2bf(float f) {
  union { __bf16 h; unsigned short s; } u;
  u.h = (__bf16)f;
  return u.s;
}
__device__ __forceinline__ unsigned pack2(unsigned short lo, unsigned short hi) {
  return (unsigned)lo | ((unsigned)hi << 16);
}
__device__ __forceinline__ v8f wmma_bf16(const Frag& a, const Frag& b, v8f c) {
  return __builtin_amdgcn_wmma_f32_16x16x32_bf16(
      /*neg_a=*/false, a.v, /*neg_b=*/false, b.v,
      /*c_mod=*/(short)0, c, /*reuse_a=*/false, /*reuse_b=*/false);
}
// A-matrix (16x32 bf16) per-lane K base for VGPR v:  k = 2v + 16*half
__device__ __forceinline__ int a_koff(int v, int half) {
  return 2 * v + (half ? 16 : 0);
}

// ---------------------------------------------------------------------------
// Kernel 0: flat fp32 -> bf16 convert, 8 elems/thread (float4 x2 -> uint4).
// ---------------------------------------------------------------------------
__global__ __launch_bounds__(256) void cvt_bf16_kernel(
    const float* __restrict__ src, unsigned short* __restrict__ dst, int n8) {
  const int i = blockIdx.x * 256 + threadIdx.x;
  if (i >= n8) return;
  const float4* s = (const float4*)src + (size_t)i * 2;
  const float4 f0 = s[0];
  const float4 f1 = s[1];
  uint4 o;
  o.x = pack2(f2bf(f0.x), f2bf(f0.y));
  o.y = pack2(f2bf(f0.z), f2bf(f0.w));
  o.z = pack2(f2bf(f1.x), f2bf(f1.y));
  o.w = pack2(f2bf(f1.z), f2bf(f1.w));
  ((uint4*)dst)[i] = o;
}

// ---------------------------------------------------------------------------
// Kernel 1: qkv = x @ w_qkv^T.  M=B*N=32768, K=512, N=1536, all bf16 in.
// Wave computes 16x64; block (8 waves) = 128x64.  grid = (256, 24).
// ---------------------------------------------------------------------------
__global__ __launch_bounds__(256) void qkv_gemm_kernel(
    const unsigned short* __restrict__ xb, const unsigned short* __restrict__ wq,
    unsigned short* __restrict__ qb, unsigned short* __restrict__ kbuf,
    unsigned short* __restrict__ vbuf) {
  const int lane = threadIdx.x & 31;
  const int wave = threadIdx.x >> 5;
  const int half = lane >> 4;
  const int lm   = lane & 15;
  const int m0 = blockIdx.x * 128 + wave * 16;   // global row in (b*n)
  const int e0 = blockIdx.y * 64;                // output feature base

  const v8f zero = {0.f, 0.f, 0.f, 0.f, 0.f, 0.f, 0.f, 0.f};
  v8f acc[4] = {zero, zero, zero, zero};

  const unsigned short* arow = xb + (size_t)(m0 + lm) * DIM;
  for (int kk = 0; kk < DIM; kk += 32) {
    __builtin_prefetch(arow + kk + 64, 0, 3);    // next-next K tile of A
    Frag a;
#pragma unroll
    for (int v = 0; v < 8; ++v)
      a.u[v] = *(const unsigned*)(arow + kk + a_koff(v, half));
#pragma unroll
    for (int nt = 0; nt < 4; ++nt) {
      // B[k][n] = w_qkv[e0+nt*16+n][k]  (row-major (e,d) weights)
      const unsigned short* brow =
          wq + (size_t)(e0 + nt * 16 + lm) * DIM + kk + 16 * half;
      Frag bw;
#pragma unroll
      for (int v = 0; v < 8; ++v)
        bw.u[v] = *(const unsigned*)(brow + 2 * v);
      acc[nt] = wmma_bf16(a, bw, acc[nt]);
    }
  }

  // Scatter to q/k/v in (b, h, n, 64) bf16 layout; fold 1/sqrt(64) into q.
#pragma unroll
  for (int nt = 0; nt < 4; ++nt) {
    const int e     = e0 + nt * 16 + lm;
    const int which = e >> 9;          // 0=q 1=k 2=v
    const int h     = (e >> 6) & 7;
    const int d     = e & 63;
    unsigned short* dst = (which == 0) ? qb : (which == 1) ? kbuf : vbuf;
    const float scl = (which == 0) ? 0.125f : 1.0f;
#pragma unroll
    for (int v = 0; v < 8; ++v) {
      const int r = m0 + v + 8 * half;
      const int b = r >> 13;           // r / 8192
      const int n = r & (NSEQ - 1);
      dst[(((size_t)b * NH + h) * NSEQ + n) * DH + d] = f2bf(acc[nt][v] * scl);
    }
  }
}

// ---------------------------------------------------------------------------
// Kernel 2: local attention, one block per (b, h, window).  256 threads.
// LDS: scores (192 KB) + v^T (48 KB) + k-tile (48 KB) = 288 KB dynamic.
// k-tile is staged once per block via async copy (ASYNCcnt); out-of-window
// rows are address-clamped -- the positional softmax mask discards them.
// ---------------------------------------------------------------------------
__global__ __launch_bounds__(256) void local_attn_kernel(
    const unsigned short* __restrict__ qb, const unsigned short* __restrict__ kbuf,
    const unsigned short* __restrict__ vbuf, unsigned short* __restrict__ ob) {
  extern __shared__ float smem[];
  unsigned short* vT = (unsigned short*)((char*)smem + LDS_VT);  // [64][384]
  unsigned short* kT = (unsigned short*)((char*)smem + LDS_KT);  // [384][64]
  const int lane = threadIdx.x & 31;
  const int wave = threadIdx.x >> 5;            // query M-tile 0..7
  const int half = lane >> 4;
  const int lm   = lane & 15;
  const int w    = blockIdx.x & (NW - 1);
  const int bh   = blockIdx.x >> 6;             // b*NH + h
  const int jBase = (w - 1) * WIN;              // key index of score column 0

  const v8f zero = {0.f, 0.f, 0.f, 0.f, 0.f, 0.f, 0.f, 0.f};

  // ---- Async-stage k tile into LDS: kT[j][d], 384 rows x 128 B.
  // 3072 x 16-byte chunks = 12 issues per thread (tracked by ASYNCcnt).
#pragma unroll
  for (int it = 0; it < 12; ++it) {
    const int idx = it * 256 + threadIdx.x;     // 0..3071
    const int j   = idx >> 3;                   // row 0..383
    const int ch  = idx & 7;                    // 16B chunk in row
    int jg = jBase + j;
    if ((unsigned)jg >= (unsigned)NSEQ) jg = 0; // clamp: masked positionally
    const unsigned short* g = kbuf + ((size_t)bh * NSEQ + jg) * DH + ch * 8;
    const unsigned lds_addr = (unsigned)(LDS_KT + j * (DH * 2) + ch * 16);
    asm volatile("global_load_async_to_lds_b128 %0, %1, off"
                 :: "v"(lds_addr), "v"(g) : "memory");
  }

  // ---- Stage v^T into LDS (coalesced dword reads, transpose b16 writes).
  for (int idx = threadIdx.x; idx < KW * (DH / 2); idx += 256) {
    const int c  = idx >> 5;          // key column 0..383
    const int dp = (idx & 31) * 2;    // even head-dim
    const int j  = jBase + c;
    const unsigned val = ((unsigned)j < (unsigned)NSEQ)
        ? *(const unsigned*)(vbuf + ((size_t)bh * NSEQ + j) * DH + dp) : 0u;
    vT[(size_t)dp * KW + c]       = (unsigned short)(val & 0xFFFFu);
    vT[(size_t)(dp + 1) * KW + c] = (unsigned short)(val >> 16);
  }

  // Pre-load both A fragments (q is pre-scaled bf16, contiguous pairs).
  const unsigned short* qrow =
      qb + ((size_t)bh * NSEQ + w * WIN + wave * 16 + lm) * DH;
  Frag aq[2];
#pragma unroll
  for (int t = 0; t < 2; ++t)
#pragma unroll
    for (int v = 0; v < 8; ++v)
      aq[t].u[v] = *(const unsigned*)(qrow + t * 32 + a_koff(v, half));

  asm volatile("s_wait_asynccnt 0" ::: "memory");
  __syncthreads();   // k tile + v^T visible to all waves

  // ---- Phase A: sim = q @ k^T  (B fragments: dword pair loads from LDS)
  for (int nt = 0; nt < 24; ++nt) {
    const unsigned short* krow = kT + (size_t)(nt * 16 + lm) * DH;
    v8f c = zero;
#pragma unroll
    for (int t = 0; t < 2; ++t) {
      Frag bk;
#pragma unroll
      for (int v = 0; v < 8; ++v)
        bk.u[v] = *(const unsigned*)(krow + t * 32 + 2 * v + 16 * half);
      c = wmma_bf16(aq[t], bk, c);
    }
#pragma unroll
    for (int v = 0; v < 8; ++v)
      smem[(size_t)(wave * 16 + v + 8 * half) * KW + nt * 16 + lm] = c[v];
  }
  __syncthreads();   // scores complete

  // ---- Phase B: masked softmax per row; re-quantize attn to bf16 in place.
  if (threadIdx.x < WIN) {
    float* rowp = smem + (size_t)threadIdx.x * KW;
    const int c0 = (jBase < 0) ? WIN : 0;              // first valid column
    const int c1 = (jBase + KW > NSEQ) ? 2 * WIN : KW; // one past last valid
    float mx = -3.402823466e38f;
    for (int c = c0; c < c1; ++c) mx = fmaxf(mx, rowp[c]);
    float sum = 0.f;
    for (int c = c0; c < c1; ++c) {
      float p = __expf(rowp[c] - mx);
      rowp[c] = p;
      sum += p;
    }
    const float inv = 1.f / sum;
    unsigned short* prow = (unsigned short*)rowp;      // overwrite low half
    for (int c = 0; c < KW; ++c) {
      const float p = (c >= c0 && c < c1) ? rowp[c] * inv : 0.f;
      prow[c] = f2bf(p);                               // slot c/2 <= c: safe
    }
  }
  __syncthreads();

  // ---- Phase C: out = attn @ v   (A + B both from LDS, dword pair loads)
  v8f oacc[4] = {zero, zero, zero, zero};
  const unsigned short* arow =
      (const unsigned short*)(smem + (size_t)(wave * 16 + lm) * KW);
  for (int ks = 0; ks < KW; ks += 32) {
    Frag aa;
#pragma unroll
    for (int v = 0; v < 8; ++v)
      aa.u[v] = *(const unsigned*)(arow + ks + a_koff(v, half));
#pragma unroll
    for (int nt = 0; nt < 4; ++nt) {
      const unsigned short* vrow = vT + (size_t)(nt * 16 + lm) * KW;
      Frag bv;
#pragma unroll
      for (int v = 0; v < 8; ++v)
        bv.u[v] = *(const unsigned*)(vrow + ks + 2 * v + 16 * half);
      oacc[nt] = wmma_bf16(aa, bv, oacc[nt]);
    }
  }

  // Store attention output in (b, n, h*64+d) bf16 layout for K3.
  const int b = bh >> 3, h = bh & 7;
#pragma unroll
  for (int nt = 0; nt < 4; ++nt) {
    const int d = nt * 16 + lm;
#pragma unroll
    for (int v = 0; v < 8; ++v) {
      const int n = w * WIN + wave * 16 + v + 8 * half;
      ob[((size_t)b * NSEQ + n) * DIM + h * DH + d] = f2bf(oacc[nt][v]);
    }
  }
}

// ---------------------------------------------------------------------------
// Kernel 3: out = o @ w_out^T.  M=32768, K=512, N=512, bf16 in, fp32 out.
// ---------------------------------------------------------------------------
__global__ __launch_bounds__(256) void out_gemm_kernel(
    const unsigned short* __restrict__ ob, const unsigned short* __restrict__ wo,
    float* __restrict__ out) {
  const int lane = threadIdx.x & 31;
  const int wave = threadIdx.x >> 5;
  const int half = lane >> 4;
  const int lm   = lane & 15;
  const int m0 = blockIdx.x * 128 + wave * 16;
  const int n0 = blockIdx.y * 64;

  const v8f zero = {0.f, 0.f, 0.f, 0.f, 0.f, 0.f, 0.f, 0.f};
  v8f acc[4] = {zero, zero, zero, zero};

  const unsigned short* arow = ob + (size_t)(m0 + lm) * DIM;
  for (int kk = 0; kk < DIM; kk += 32) {
    __builtin_prefetch(arow + kk + 64, 0, 3);
    Frag a;
#pragma unroll
    for (int v = 0; v < 8; ++v)
      a.u[v] = *(const unsigned*)(arow + kk + a_koff(v, half));
#pragma unroll
    for (int nt = 0; nt < 4; ++nt) {
      const unsigned short* brow =
          wo + (size_t)(n0 + nt * 16 + lm) * DIM + kk + 16 * half;
      Frag bw;
#pragma unroll
      for (int v = 0; v < 8; ++v)
        bw.u[v] = *(const unsigned*)(brow + 2 * v);
      acc[nt] = wmma_bf16(a, bw, acc[nt]);
    }
  }
#pragma unroll
  for (int nt = 0; nt < 4; ++nt)
#pragma unroll
    for (int v = 0; v < 8; ++v)
      out[(size_t)(m0 + v + 8 * half) * DIM + n0 + nt * 16 + lm] = acc[nt][v];
}

// ---------------------------------------------------------------------------
extern "C" void kernel_launch(void* const* d_in, const int* in_sizes, int n_in,
                              void* d_out, int out_size, void* d_ws, size_t ws_size,
                              hipStream_t stream) {
  (void)in_sizes; (void)n_in; (void)out_size; (void)ws_size;
  const float* x    = (const float*)d_in[0];   // (4, 8192, 512) f32
  const float* wqkv = (const float*)d_in[1];   // (1536, 512)    f32
  const float* wout = (const float*)d_in[2];   // (512, 512)     f32
  float* out = (float*)d_out;                  // (4, 8192, 512) f32

  // Workspace (bf16 ushorts): q,k,v,o (b,h,n,64)/(b,n,512) + xb + weights.
  const size_t elems = (size_t)B_ * NSEQ * DIM;   // 16,777,216
  unsigned short* qb   = (unsigned short*)d_ws;
  unsigned short* kb   = qb + elems;
  unsigned short* vb   = kb + elems;
  unsigned short* obuf = vb + elems;
  unsigned short* xb   = obuf + elems;
  unsigned short* wqb  = xb + elems;
  unsigned short* wob  = wqb + (size_t)3 * DIM * DIM;   // 786,432
  // total: 5*16,777,216 + 786,432 + 262,144 ushorts  ~= 170 MB

  (void)hipFuncSetAttribute(reinterpret_cast<const void*>(local_attn_kernel),
                            hipFuncAttributeMaxDynamicSharedMemorySize,
                            LDS_TOTAL);

  const int nx8 = (int)(elems / 8);
  const int nq8 = 3 * DIM * DIM / 8;
  const int no8 = DIM * DIM / 8;
  cvt_bf16_kernel<<<(nx8 + 255) / 256, 256, 0, stream>>>(x, xb, nx8);
  cvt_bf16_kernel<<<(nq8 + 255) / 256, 256, 0, stream>>>(wqkv, wqb, nq8);
  cvt_bf16_kernel<<<(no8 + 255) / 256, 256, 0, stream>>>(wout, wob, no8);

  qkv_gemm_kernel<<<dim3(256, 24), 256, 0, stream>>>(xb, wqb, qb, kb, vb);
  local_attn_kernel<<<dim3(B_ * NH * NW), 256, LDS_TOTAL, stream>>>(qb, kb, vb, obuf);
  out_gemm_kernel<<<dim3(256, 8), 256, 0, stream>>>(obuf, wob, out);
}
